// NymstromAttention_76562087018656
// MI455X (gfx1250) — matrix operand order repaired and checked by reading, not compiled, not run on hardware
//
#include <hip/hip_runtime.h>
#include <hip/hip_bf16.h>
#include <stdint.h>

#define DEV __device__ __forceinline__

typedef __attribute__((ext_vector_type(16))) __bf16 v16bf;
typedef __attribute__((ext_vector_type(8)))  float  v8f;

// Problem constants (fixed by the reference).
constexpr int  Bc  = 2;
constexpr int  Lc  = 4096;
constexpr int  Dc  = 1024;
constexpr int  Hc  = 16;
constexpr int  DHc = 64;
constexpr int  Mc  = 128;   // bottleneck / landmarks
constexpr float EPSc = 1e-4f;

// ---------------- bf16 helpers (stored as ushort, RNE conversion) ----------
DEV unsigned short f2bf(float f) {
  unsigned u = __float_as_uint(f);
  u += 0x7FFFu + ((u >> 16) & 1u);
  return (unsigned short)(u >> 16);
}
DEV float bf2f(unsigned short h) { return __uint_as_float(((unsigned)h) << 16); }

// ---------------- CDNA5 tile staging: TDM (preferred) or async-to-LDS ------
constexpr int BM = 128, BN = 128, BK = 32;
constexpr int BKP = 40;  // LDS row stride: 80B = 16B-aligned AND 16-row conflict-free

#if __has_builtin(__builtin_amdgcn_tensor_load_to_lds)
#define HAVE_TDM 1
typedef unsigned int v4u __attribute__((ext_vector_type(4)));
typedef int          v8i __attribute__((ext_vector_type(8)));
typedef int          v4i __attribute__((ext_vector_type(4)));

// One TDM descriptor = one 128x32 bf16 tile, global->LDS, with LDS padding of
// 4 DWORDs after every 16 DWORDs (i.e. 64B rows -> 80B LDS stride == BKP).
// tensor_dim1 < tile rows gives hardware zero-fill for the N=64 launches.
DEV void tdm_tile_2d(unsigned lds_addr, const void* gaddr,
                     unsigned tdim0, unsigned tdim1,
                     unsigned tile0, unsigned tile1, unsigned stride0) {
  unsigned long long ga = (unsigned long long)gaddr;
  v4u g0 = { 1u,                                  // count=1 (valid), user mode
             lds_addr,                            // D#.lds_addr (bytes)
             (unsigned)(ga & 0xFFFFFFFFu),        // global_addr[31:0]
             (unsigned)(ga >> 32) | (2u << 30) }; // global_addr[56:32] | type=2
  v8i g1 = {
    (int)((1u << 16) | (1u << 20) | (3u << 22) | (3u << 25)), // data_size=2B,
                                                 // pad_enable, interval=16DW, amount=4DW
    (int)((tdim0 & 0xFFFFu) << 16),              // tensor_dim0[15:0] @ bits 63:48
    (int)((tdim0 >> 16) | ((tdim1 & 0xFFFFu) << 16)), // dim0[31:16] | dim1[15:0]
    (int)((tdim1 >> 16) | (tile0 << 16)),        // dim1[31:16] | tile_dim0
    (int)tile1,                                  // tile_dim1 (tile_dim2=0 -> 2D)
    (int)stride0,                                // tensor_dim0_stride[31:0]
    0, 0 };
  v4i zz4 = {0, 0, 0, 0};
  v8i zz8 = {0, 0, 0, 0, 0, 0, 0, 0};
  // 6-arg form (therock-10.0 HIP headers / clang-23 toolchain).
  __builtin_amdgcn_tensor_load_to_lds(g0, g1, zz4, zz4, zz8, 0);
}
DEV void wait_tiles() { __builtin_amdgcn_s_wait_tensorcnt(0); }
#else
#define HAVE_TDM 0
// Fallback: per-lane GLOBAL_LOAD_ASYNC_TO_LDS_B128 (ASYNCcnt-tracked).
DEV void async_b128(unsigned ldsOff, const void* gaddr) {
  asm volatile("global_load_async_to_lds_b128 %0, %1, off"
               :: "v"(ldsOff), "v"(gaddr) : "memory");
}
DEV void wait_tiles() {
#if __has_builtin(__builtin_amdgcn_s_wait_asynccnt)
  __builtin_amdgcn_s_wait_asynccnt(0);
#else
  asm volatile("s_wait_asynccnt 0" ::: "memory");
#endif
}
#endif

// ---------------- WMMA GEMM engine -----------------------------------------
// Block tile 128x128, K-step 32 (== one v_wmma_f32_16x16x32_bf16 K).
// 256 threads = 8 waves arranged 2x4; each wave owns a 64x32 sub-tile
// (4x2 fragments, 8 WMMAs per K-step). Both A and B are consumed as
// [rows][K] bf16 row-major ("N-major" for B), double-buffered in LDS.
struct GemmP {
  const unsigned short* A;   // bf16 [M x K] row-major (per-batch stride sA)
  const unsigned short* B;   // bf16 [N x K] row-major (per-batch stride sB)
  float*           Cf;       // EPI 0
  unsigned short*  Cb;       // EPI 1..6
  int M, N, K;
  int lda, ldb, ldc;
  long sA, sB, sC;
  int aMod, bMod;            // if >0, batch index for A/B is z % mod (per-head operands)
  const float* bias;         // EPI 0/1
  const int*   mask;         // EPI 2 (nys_K key mask), may be null
  const float* norm;         // EPI 4 (per-row normalizer)
};

union Frag { v16bf v; unsigned u[8]; };

// A-matrix 16x32 bf16 fragment: lanes 0-15 row M=lane, K=0..15;
// lanes 16-31 row M=lane-16, K=16..31 (contiguous K, matches frag_b).  Two b128 LDS loads.
DEV v16bf frag_a(const unsigned short* base, int lane) {
  int row = lane & 15;
  int kh  = (lane >> 4) << 4;            // 0 or 16
  const unsigned short* p = base + row * BKP + kh;
  uint4 a = *(const uint4*)(p);
  uint4 b = *(const uint4*)(p + 8);
  Frag f;
  f.u[0] = a.x; f.u[1] = a.y; f.u[2] = a.z; f.u[3] = a.w;
  f.u[4] = b.x; f.u[5] = b.y; f.u[6] = b.z; f.u[7] = b.w;
  return f.v;
}

// B-matrix 32x16 bf16 fragment: lane holds column N=lane&15;
// lanes 0-15 K=0..15, lanes 16-31 K=16..31 (contiguous K per lane).
DEV v16bf frag_b(const unsigned short* base, int lane) {
  int n  = lane & 15;
  int kb = (lane >> 4) << 4;             // 0 or 16
  const unsigned short* p = base + n * BKP + kb;
  uint4 a = *(const uint4*)(p);
  uint4 b = *(const uint4*)(p + 8);
  Frag f;
  f.u[0] = a.x; f.u[1] = a.y; f.u[2] = a.z; f.u[3] = a.w;
  f.u[4] = b.x; f.u[5] = b.y; f.u[6] = b.z; f.u[7] = b.w;
  return f.v;
}

template <int EPI>
__global__ __launch_bounds__(256) void gemm_bf16(GemmP p) {
  __shared__ __align__(16) unsigned short As[2][BM * BKP];
  __shared__ __align__(16) unsigned short Bs[2][BN * BKP];

  const int tid   = threadIdx.x;
  const int lane  = tid & 31;
  const int wave  = tid >> 5;
  const int waveM = wave >> 2;           // 0..1  -> 64-row slab
  const int waveN = wave & 3;            // 0..3  -> 32-col slab
  const int z  = blockIdx.z;
  const int m0 = blockIdx.y * BM;
  const int n0 = blockIdx.x * BN;

  const unsigned short* Ag = p.A + (long)(p.aMod ? (z % p.aMod) : z) * p.sA;
  const unsigned short* Bg = p.B + (long)(p.bMod ? (z % p.bMod) : z) * p.sB;

  const unsigned aBase[2] = {(unsigned)(uintptr_t)&As[0][0], (unsigned)(uintptr_t)&As[1][0]};
  const unsigned bBase[2] = {(unsigned)(uintptr_t)&Bs[0][0], (unsigned)(uintptr_t)&Bs[1][0]};

  // Issue one k-step's worth of tile copies into LDS buffer `buf`.
  auto stage = [&](int k0, int buf) {
#if HAVE_TDM
    if (wave == 0) {
      // A tile: rows always in-range (M is a multiple of 128 for every launch).
      tdm_tile_2d(aBase[buf], Ag + (long)m0 * p.lda + k0,
                  (unsigned)(p.K - k0), (unsigned)BM, BK, BM, (unsigned)p.lda);
      // B tile: tensor_dim1 = remaining rows -> TDM zero-fills rows >= N.
      tdm_tile_2d(bBase[buf], Bg + (long)n0 * p.ldb + k0,
                  (unsigned)(p.K - k0), (unsigned)(p.N - n0), BK, BN, (unsigned)p.ldb);
    }
#else
#pragma unroll
    for (int t = 0; t < 2; ++t) {
      int idx = t * 256 + tid;
      int r = idx >> 2;                  // 0..127
      int c = (idx & 3) << 3;            // 0,8,16,24 elements (16B chunks)
      unsigned loff = (unsigned)((r * BKP + c) * 2);
      async_b128(aBase[buf] + loff, Ag + (long)(m0 + r) * p.lda + (k0 + c));
      int gn = n0 + r;
      int gc = gn < p.N ? gn : p.N - 1;  // branchless clamp for N=64 launches
      async_b128(bBase[buf] + loff, Bg + (long)gc * p.ldb + (k0 + c));
    }
#endif
  };

  v8f acc[4][2];
#pragma unroll
  for (int i = 0; i < 4; ++i)
#pragma unroll
    for (int j = 0; j < 2; ++j)
#pragma unroll
      for (int e = 0; e < 8; ++e) acc[i][j][e] = 0.0f;

  const int nsteps = p.K / BK;
  stage(0, 0);
  for (int s = 0; s < nsteps; ++s) {
    const int cur = s & 1;
#if HAVE_TDM
    if (wave == 0) wait_tiles();         // only wave 0 issued the DMAs
#else
    wait_tiles();
#endif
    __syncthreads();                     // stage(s) visible to all; prev compute done
    if (s + 1 < nsteps) stage((s + 1) * BK, cur ^ 1);

    v16bf af[4], bfr[2];
#pragma unroll
    for (int i = 0; i < 4; ++i)
      af[i] = frag_a(&As[cur][(waveM * 64 + i * 16) * BKP], lane);
#pragma unroll
    for (int j = 0; j < 2; ++j)
      bfr[j] = frag_b(&Bs[cur][(waveN * 32 + j * 16) * BKP], lane);
#pragma unroll
    for (int i = 0; i < 4; ++i)
#pragma unroll
      for (int j = 0; j < 2; ++j)
        acc[i][j] = __builtin_amdgcn_wmma_f32_16x16x32_bf16(
            false, af[i], false, bfr[j], (short)0, acc[i][j], false, false);
  }

  // ---- epilogue: lanes 0-15 N=lane, VGPR r -> M=r; lanes 16-31 M=r+8
  const int colf = lane & 15;
  const int rowh = (lane >> 4) << 3;
#pragma unroll
  for (int i = 0; i < 4; ++i) {
#pragma unroll
    for (int j = 0; j < 2; ++j) {
      int cB = n0 + waveN * 32 + j * 16 + colf;
      if (cB >= p.N) continue;
#pragma unroll
      for (int r = 0; r < 8; ++r) {
        int rB = m0 + waveM * 64 + i * 16 + rowh + r;
        float v = acc[i][j][r];
        if (EPI == 0) {                  // f32 linear + bias (final projection)
          float bb = p.bias ? p.bias[cB] : 0.0f;
          p.Cf[(long)z * p.sC + (long)rB * p.ldc + cB] = v + bb;
        } else if (EPI == 1) {           // bf16, split heads [B,H,L,dh]
          float bb = p.bias ? p.bias[cB] : 0.0f;
          int b_ = rB >> 12, l_ = rB & (Lc - 1);
          int h_ = cB >> 6,  d_ = cB & (DHc - 1);
          p.Cb[(((long)b_ * Hc + h_) * Lc + l_) * DHc + d_] = f2bf(v + bb);
        } else if (EPI == 2) {           // exp (+ optional key mask on column)
          float e = __expf(v);
          if (p.mask) {
            int b_ = z >> 4;             // z / H
            e = p.mask[(long)b_ * Lc + cB] ? e : 0.0f;
          }
          p.Cb[(long)z * p.sC + (long)rB * p.ldc + cB] = f2bf(e);
        } else if (EPI == 3) {           // plain bf16 linear (adj_Q)
          p.Cb[(long)z * p.sC + (long)rB * p.ldc + cB] = f2bf(v);
        } else if (EPI == 4) {           // divide by norm, join heads [B,L,H*dh]
          float nv = p.norm[(long)z * Lc + rB];
          float hv = v / nv;
          int b_ = z >> 4, h_ = z & (Hc - 1);
          p.Cb[(((long)b_ * Lc + rB) * Dc) + (long)h_ * DHc + cB] = f2bf(hv);
        } else if (EPI == 5) {           // bf16, transposed split heads [B,H,dh,L] (V)
          float bb = p.bias ? p.bias[cB] : 0.0f;
          int b_ = rB >> 12, l_ = rB & (Lc - 1);
          int h_ = cB >> 6,  d_ = cB & (DHc - 1);
          p.Cb[(((long)b_ * Hc + h_) * DHc + d_) * Lc + l_] = f2bf(v + bb);
        } else {                         // EPI 6: bf16 transposed linear (KV^T)
          p.Cb[(long)z * p.sC + (long)cB * p.ldc + rB] = f2bf(v);
        }
      }
    }
  }
}

// ---------------- per-head Gram + in-place Gauss-Jordan inverse ------------
// A = exp(bot_h bot_h^T) + eps*I (128x128 f32 in 66KB of the 320KB WGP LDS).
// Writes Ainv TRANSPOSED (N-major) for direct consumption by the GEMM engine.
__global__ __launch_bounds__(256) void nysmat_invert(
    const float* __restrict__ bot, unsigned short* __restrict__ ainvT) {
  constexpr int LD = Mc + 1;
  __shared__ float Amat[Mc * LD];
  __shared__ float fac[Mc];
  const int h = blockIdx.x;
  const int tid = threadIdx.x;
  const float* bh = bot + (long)h * Mc * DHc;

  for (int e = tid; e < Mc * Mc; e += 256) {
    int n = e >> 7, m = e & (Mc - 1);
    const float* pn = bh + n * DHc;
    const float* pm = bh + m * DHc;
    float s = 0.0f;
#pragma unroll 8
    for (int d = 0; d < DHc; ++d) s += pn[d] * pm[d];
    float val = __expf(s);
    if (n == m) val += EPSc;
    Amat[n * LD + m] = val;
  }
  __syncthreads();

  for (int k = 0; k < Mc; ++k) {
    for (int i = tid; i < Mc; i += 256) fac[i] = Amat[i * LD + k];
    __syncthreads();
    float ip = 1.0f / fac[k];
    for (int c = tid; c < Mc; c += 256)
      Amat[k * LD + c] = (c == k) ? ip : Amat[k * LD + c] * ip;
    __syncthreads();
    for (int e = tid; e < Mc * Mc; e += 256) {
      int i = e >> 7, c = e & (Mc - 1);
      if (i == k) continue;
      float f = fac[i];
      Amat[i * LD + c] = (c == k) ? (-f * ip) : (Amat[i * LD + c] - f * Amat[k * LD + c]);
    }
    __syncthreads();
  }

  // ainvT[n][k] = Ainv[k][n]
  for (int e = tid; e < Mc * Mc; e += 256) {
    int n = e >> 7, k = e & (Mc - 1);
    ainvT[(long)h * Mc * Mc + e] = f2bf(Amat[k * LD + n]);
  }
}

// ---------------- small kernels --------------------------------------------
__global__ void cast_f32_bf16(const float* __restrict__ in,
                              unsigned short* __restrict__ out, long n) {
  long i = (long)blockIdx.x * blockDim.x + threadIdx.x;
  if (i < n) out[i] = f2bf(in[i]);
}

// out[n][k] = bf16(in[k][n]); LDS-tiled transpose-cast for the weights.
__global__ __launch_bounds__(256) void transpose_cast(
    const float* __restrict__ in, unsigned short* __restrict__ out,
    int rows, int cols) {
  __shared__ float tile[32][33];
  int bx = blockIdx.x * 32;              // col (n) base
  int by = blockIdx.y * 32;              // row (k) base
  int tx = threadIdx.x & 31, ty = threadIdx.x >> 5;
  for (int r = ty; r < 32; r += 8)
    tile[r][tx] = in[(long)(by + r) * cols + bx + tx];
  __syncthreads();
  for (int r = ty; r < 32; r += 8)
    out[(long)(bx + r) * rows + by + tx] = f2bf(tile[tx][r]);
}

// ksum[z][m] = sum_l nysK[z][m][l]; one wave32 per row, shuffle reduce.
__global__ __launch_bounds__(256) void ksum_kernel(
    const unsigned short* __restrict__ nysK, float* __restrict__ ksum) {
  int g = blockIdx.x * blockDim.x + threadIdx.x;
  int w = g >> 5, lane = g & 31;
  if (w >= Bc * Hc * Mc) return;
  const unsigned short* rowp = nysK + (long)w * Lc;
  float s = 0.0f;
  for (int l = lane; l < Lc; l += 32) s += bf2f(rowp[l]);
#pragma unroll
  for (int o = 16; o > 0; o >>= 1) s += __shfl_xor(s, o, 32);
  if (lane == 0) ksum[w] = s;
}

// norm[z][l] = dot(adjQ[z][l][:], ksum[z][:]) + eps
__global__ __launch_bounds__(256) void norm_kernel(
    const unsigned short* __restrict__ adjQ, const float* __restrict__ ksum,
    float* __restrict__ norm) {
  long i = (long)blockIdx.x * blockDim.x + threadIdx.x;
  if (i >= (long)Bc * Hc * Lc) return;
  int z = (int)(i >> 12);                // / L
  const unsigned short* a = adjQ + i * Mc;
  const float* ks = ksum + (long)z * Mc;
  float s = EPSc;
#pragma unroll 8
  for (int m = 0; m < Mc; ++m) s += bf2f(a[m]) * ks[m];
  norm[i] = s;
}

// ---------------- host side -------------------------------------------------
extern "C" void kernel_launch(void* const* d_in, const int* in_sizes, int n_in,
                              void* d_out, int out_size, void* d_ws, size_t ws_size,
                              hipStream_t stream) {
  (void)in_sizes; (void)n_in; (void)out_size; (void)ws_size;
  const float* queries    = (const float*)d_in[0];
  const int*   attn_mask  = (const int*)d_in[1];
  const float* bottleneck = (const float*)d_in[2];
  const float* W_q = (const float*)d_in[3];
  const float* b_q = (const float*)d_in[4];
  const float* W_k = (const float*)d_in[5];
  const float* b_k = (const float*)d_in[6];
  const float* W_v = (const float*)d_in[7];
  const float* b_v = (const float*)d_in[8];
  const float* W_o = (const float*)d_in[9];
  const float* b_o = (const float*)d_in[10];
  float* out = (float*)d_out;

  char* ws = (char*)d_ws;
  size_t off = 0;
  auto alloc = [&](size_t bytes) {
    char* p = ws + off;
    off += (bytes + 255) & ~(size_t)255;
    return p;
  };
  const long nBLD = (long)Bc * Lc * Dc;
  unsigned short* qbf   = (unsigned short*)alloc(nBLD * 2);
  unsigned short* wqbf  = (unsigned short*)alloc((long)Dc * Dc * 2);  // W^T layouts
  unsigned short* wkbf  = (unsigned short*)alloc((long)Dc * Dc * 2);
  unsigned short* wvbf  = (unsigned short*)alloc((long)Dc * Dc * 2);
  unsigned short* wobf  = (unsigned short*)alloc((long)Dc * Dc * 2);
  unsigned short* botbf = (unsigned short*)alloc((long)Hc * Mc * DHc * 2);
  unsigned short* Qs    = (unsigned short*)alloc(nBLD * 2);   // [B,H,L,dh]
  unsigned short* Ks    = (unsigned short*)alloc(nBLD * 2);   // [B,H,L,dh]
  unsigned short* Vt    = (unsigned short*)alloc(nBLD * 2);   // [B,H,dh,L]
  unsigned short* ainvT = (unsigned short*)alloc((long)Hc * Mc * Mc * 2);
  unsigned short* nysQ  = (unsigned short*)alloc((long)Bc * Hc * Lc * Mc * 2);
  unsigned short* nysK  = (unsigned short*)alloc((long)Bc * Hc * Mc * Lc * 2);
  unsigned short* adjQ  = (unsigned short*)alloc((long)Bc * Hc * Lc * Mc * 2);
  unsigned short* kvT   = (unsigned short*)alloc((long)Bc * Hc * DHc * Mc * 2); // [z][dh][M]
  unsigned short* joined= (unsigned short*)alloc(nBLD * 2);   // [B,L,D]
  float* ksum  = (float*)alloc((long)Bc * Hc * Mc * 4);
  float* normb = (float*)alloc((long)Bc * Hc * Lc * 4);

  // 1) casts: activations plain, weights transposed to [N][K]
  cast_f32_bf16<<<dim3((unsigned)((nBLD + 255) / 256)), 256, 0, stream>>>(queries, qbf, nBLD);
  dim3 tg(Dc / 32, Dc / 32);
  transpose_cast<<<tg, 256, 0, stream>>>(W_q, wqbf, Dc, Dc);
  transpose_cast<<<tg, 256, 0, stream>>>(W_k, wkbf, Dc, Dc);
  transpose_cast<<<tg, 256, 0, stream>>>(W_v, wvbf, Dc, Dc);
  transpose_cast<<<tg, 256, 0, stream>>>(W_o, wobf, Dc, Dc);
  cast_f32_bf16<<<dim3((Hc * Mc * DHc) / 256), 256, 0, stream>>>(
      bottleneck, botbf, (long)Hc * Mc * DHc);

  // 2) per-head landmark Gram + inverse (writes Ainv^T)
  nysmat_invert<<<dim3(Hc), 256, 0, stream>>>(bottleneck, ainvT);

  // 3) Q/K/V projections: [B*L,D] @ W^T[N][K] -> split heads
  {
    GemmP p{}; p.A = qbf;
    p.M = Bc * Lc; p.N = Dc; p.K = Dc; p.lda = Dc; p.ldb = Dc;
    dim3 g(Dc / BN, (Bc * Lc) / BM, 1);
    p.B = wqbf; p.Cb = Qs; p.bias = b_q; gemm_bf16<1><<<g, 256, 0, stream>>>(p);
    p.B = wkbf; p.Cb = Ks; p.bias = b_k; gemm_bf16<1><<<g, 256, 0, stream>>>(p);
    p.B = wvbf; p.Cb = Vt; p.bias = b_v; gemm_bf16<5><<<g, 256, 0, stream>>>(p);
  }

  // 4) nys_Q = exp(Q @ bot^T): per (b,h): [L,64] x bot[N=128][K=64]
  {
    GemmP p{}; p.A = Qs; p.B = botbf; p.Cb = nysQ;
    p.M = Lc; p.N = Mc; p.K = DHc; p.lda = DHc; p.ldb = DHc; p.ldc = Mc;
    p.sA = (long)Lc * DHc; p.sB = (long)Mc * DHc; p.sC = (long)Lc * Mc;
    p.bMod = Hc;
    dim3 g(1, Lc / BM, Bc * Hc);
    gemm_bf16<2><<<g, 256, 0, stream>>>(p);
  }

  // 5) nys_K = mask * exp(bot @ K^T): per (b,h): [128,64] x K[N=L][K=64]
  {
    GemmP p{}; p.A = botbf; p.B = Ks; p.Cb = nysK;
    p.M = Mc; p.N = Lc; p.K = DHc; p.lda = DHc; p.ldb = DHc; p.ldc = Lc;
    p.sA = (long)Mc * DHc; p.aMod = Hc;
    p.sB = (long)Lc * DHc; p.sC = (long)Mc * Lc;
    p.mask = attn_mask;
    dim3 g(Lc / BN, 1, Bc * Hc);
    gemm_bf16<2><<<g, 256, 0, stream>>>(p);
  }

  // 6) ksum[z][m] = sum_l nysK
  ksum_kernel<<<dim3((Bc * Hc * Mc * 32) / 256), 256, 0, stream>>>(nysK, ksum);

  // 7) adj_Q = nys_Q @ Ainv: per (b,h): [L,128] x AinvT[N=128][K=128]
  {
    GemmP p{}; p.A = nysQ; p.B = ainvT; p.Cb = adjQ;
    p.M = Lc; p.N = Mc; p.K = Mc; p.lda = Mc; p.ldb = Mc; p.ldc = Mc;
    p.sA = (long)Lc * Mc; p.sB = (long)Mc * Mc; p.bMod = Hc; p.sC = (long)Lc * Mc;
    dim3 g(1, Lc / BM, Bc * Hc);
    gemm_bf16<3><<<g, 256, 0, stream>>>(p);
  }

  // 8) norm[z][l] = adj_Q . ksum + eps
  norm_kernel<<<dim3((Bc * Hc * Lc) / 256), 256, 0, stream>>>(adjQ, ksum, normb);

  // 9) KV = nys_K @ V: per (b,h): [128,L] x Vt[N=64][K=L] -> store KV^T [64][128]
  {
    GemmP p{}; p.A = nysK; p.B = Vt; p.Cb = kvT;
    p.M = Mc; p.N = DHc; p.K = Lc; p.lda = Lc; p.ldb = Lc; p.ldc = Mc;
    p.sA = (long)Mc * Lc; p.sB = (long)DHc * Lc; p.sC = (long)DHc * Mc;
    dim3 g(1, 1, Bc * Hc);
    gemm_bf16<6><<<g, 256, 0, stream>>>(p);
  }

  // 10) un = adj_Q @ KV: per (b,h): [L,128] x KVt[N=64][K=128]; /norm; join heads
  {
    GemmP p{}; p.A = adjQ; p.B = kvT; p.Cb = joined; p.norm = normb;
    p.M = Lc; p.N = DHc; p.K = Mc; p.lda = Mc; p.ldb = Mc;
    p.sA = (long)Lc * Mc; p.sB = (long)DHc * Mc;
    dim3 g(1, Lc / BM, Bc * Hc);
    gemm_bf16<4><<<g, 256, 0, stream>>>(p);
  }

  // 11) out = joined @ W_o^T + b_o (f32)
  {
    GemmP p{}; p.A = joined; p.B = wobf; p.Cf = out;
    p.M = Bc * Lc; p.N = Dc; p.K = Dc; p.lda = Dc; p.ldb = Dc; p.ldc = Dc;
    p.bias = b_o;
    dim3 g(Dc / BN, (Bc * Lc) / BM, 1);
    gemm_bf16<0><<<g, 256, 0, stream>>>(p);
  }
}
